// CrossAttentionBlock_54795192762573
// MI455X (gfx1250) — matrix operand order repaired and checked by reading, not compiled, not run on hardware
//
#include <hip/hip_runtime.h>
#include <stdint.h>

// ---------------------------------------------------------------------------
// CDNA5 (gfx1250) cross-attention block: bf16 WMMA everywhere, double-buffered
// async-to-LDS staging (saddr form), wave32 layouts per CDNA5 ISA.
// ---------------------------------------------------------------------------

typedef __attribute__((ext_vector_type(16))) __bf16 v16bf;
typedef __attribute__((ext_vector_type(8)))  float  v8f;

union FragA { unsigned int u[8]; v16bf v; };
static_assert(sizeof(v16bf) == 32, "v16bf must be 8 dwords");

__device__ __forceinline__ unsigned short f2bf(float f) {
    unsigned int u = __float_as_uint(f);
    unsigned int r = u + 0x7FFFu + ((u >> 16) & 1u);   // RNE
    return (unsigned short)(r >> 16);
}

__device__ __forceinline__ v8f wmma_bf16(const FragA& a, const FragA& b, v8f c) {
    return __builtin_amdgcn_wmma_f32_16x16x32_bf16(
        /*neg_a=*/false, a.v, /*neg_b=*/false, b.v,
        /*c_mod=*/(short)0, c, /*reuse_a=*/false, /*reuse_b=*/false);
}

// CDNA5 async copy, saddr form: uniform 64-bit base in SGPRs + 32-bit
// per-lane byte offset. 16B/lane global -> LDS, tracked by ASYNCcnt.
__device__ __forceinline__ void async_b128_s(void* lds, unsigned long long sbase,
                                             unsigned voff) {
    unsigned off = (unsigned)(uintptr_t)lds;              // AS3 offset = low 32b
    asm volatile("global_load_async_to_lds_b128 %0, %1, %2"
                 :: "v"(off), "v"(voff), "s"(sbase) : "memory");
}
__device__ __forceinline__ void wait_async() {
    asm volatile("s_wait_asynccnt 0" ::: "memory");
}
__device__ __forceinline__ void wait_ds() {
    asm volatile("s_wait_dscnt 0" ::: "memory");
}

// ---------------------------------------------------------------------------
// fp32 -> bf16 conversion (one bandwidth pass, vectorized; n % 4 == 0)
// ---------------------------------------------------------------------------
__global__ __launch_bounds__(256) void xattn_cvt_bf16(
    const float4* __restrict__ in, ushort4* __restrict__ out, int n4) {
    int i = blockIdx.x * blockDim.x + threadIdx.x;
    int stride = gridDim.x * blockDim.x;
    for (; i < n4; i += stride) {
        float4 f = in[i];
        ushort4 o;
        o.x = f2bf(f.x); o.y = f2bf(f.y); o.z = f2bf(f.z); o.w = f2bf(f.w);
        out[i] = o;
    }
}

// ---------------------------------------------------------------------------
// bf16 GEMM: out[M,Nn] = A[M,Kd] * Bw[Kd,Nn] + bias.
// Block tile 128x128, BK=32; 8 waves, each 32x64 via 2x4 WMMA accumulators.
// Double-buffered LDS: async-prefetch tile i+1 while WMMAs consume tile i.
// If outF != nullptr: out = resid + sigmoid(gate)*(acc+bias)  (fp32 epilogue).
// ---------------------------------------------------------------------------
__global__ __launch_bounds__(256) void xattn_gemm_bf16(
    const unsigned short* __restrict__ A,
    const unsigned short* __restrict__ Bw,
    const float* __restrict__ bias,
    unsigned short* __restrict__ outBf,
    float* __restrict__ outF,
    const float* __restrict__ resid,
    const float* __restrict__ gate,
    int M, int Kd, int Nn)
{
    __shared__ unsigned short As[2][128 * 40];   // row-major [128][32+8]
    __shared__ unsigned short Bt[2][128 * 40];   // transposed [Ncol][32+8]

    const int tid  = threadIdx.x;
    const int wid  = tid >> 5, lane = tid & 31;
    const int half = lane >> 4, l16 = lane & 15;
    const int wm   = wid >> 1,  wn  = wid & 1;
    const int blockN = blockIdx.x * 128;
    const int blockM = blockIdx.y * 128;

    const v8f vzero = {};
    v8f acc[2][4];
    #pragma unroll
    for (int a = 0; a < 2; a++)
        #pragma unroll
        for (int b = 0; b < 4; b++) acc[a][b] = vzero;

    // --- staging helpers (2 transfers per thread each, fully unrolled) ---
    auto stageA = [&](int buf, int kb) {
        unsigned long long base =
            (unsigned long long)(uintptr_t)(A + (size_t)blockM * Kd + kb);
        const int row0 = tid >> 2, seg = tid & 3;       // (tid+256)&3 == tid&3
        async_b128_s(&As[buf][row0 * 40 + seg * 8], base,
                     (unsigned)((row0 * Kd + seg * 8) * 2));
        const int row1 = row0 + 64;
        async_b128_s(&As[buf][row1 * 40 + seg * 8], base,
                     (unsigned)((row1 * Kd + seg * 8) * 2));
    };
    auto stageB = [&](int buf, int kb) {
        #pragma unroll
        for (int t = 0; t < 2; ++t) {
            int q = tid + t * 256;
            int kr = q >> 4, seg = q & 15;
            uint4 d = *(const uint4*)(Bw + (size_t)(kb + kr) * Nn + blockN + seg * 8);
            unsigned int w[4] = {d.x, d.y, d.z, d.w};
            #pragma unroll
            for (int e = 0; e < 4; e++) {
                int c = seg * 8 + e * 2;
                Bt[buf][(c + 0) * 40 + kr] = (unsigned short)(w[e] & 0xFFFFu);
                Bt[buf][(c + 1) * 40 + kr] = (unsigned short)(w[e] >> 16);
            }
        }
    };

    const int nk = Kd >> 5;
    stageA(0, 0);
    stageB(0, 0);

    for (int i = 0; i < nk; ++i) {
        const int p = i & 1;
        wait_async();
        __syncthreads();                 // buf[p] ready; buf[1-p] free to refill
        if (i + 1 < nk) { stageA(1 - p, (i + 1) << 5); stageB(1 - p, (i + 1) << 5); }

        const unsigned int* As32 = (const unsigned int*)As[p];
        const unsigned int* Bt32 = (const unsigned int*)Bt[p];
        FragA fa[2], fb[4];
        #pragma unroll
        for (int ms = 0; ms < 2; ms++) {
            int row = wm * 32 + ms * 16 + l16;
            #pragma unroll
            for (int j = 0; j < 8; j++) {
                int pair = (j < 4) ? (j + 4 * half) : (8 + (j - 4) + 4 * half);
                fa[ms].u[j] = As32[row * 20 + pair];
            }
        }
        #pragma unroll
        for (int nt = 0; nt < 4; nt++) {
            int col = wn * 64 + nt * 16 + l16;
            #pragma unroll
            for (int j = 0; j < 8; j++)
                fb[nt].u[j] = Bt32[col * 20 + ((j < 4) ? (j + 4 * half) : (4 + j + 4 * half))];
        }
        #pragma unroll
        for (int ms = 0; ms < 2; ms++)
            #pragma unroll
            for (int nt = 0; nt < 4; nt++)
                acc[ms][nt] = wmma_bf16(fa[ms], fb[nt], acc[ms][nt]);
    }

    float g = 0.f;
    if (outF) { float gv = gate[0]; g = 1.f / (1.f + __expf(-gv)); }
    #pragma unroll
    for (int ms = 0; ms < 2; ms++) {
        #pragma unroll
        for (int nt = 0; nt < 4; nt++) {
            int colg = blockN + wn * 64 + nt * 16 + l16;
            float bv = bias[colg];
            #pragma unroll
            for (int j = 0; j < 8; j++) {
                int rowg = blockM + wm * 32 + ms * 16 + j + 8 * half;
                float val = acc[ms][nt][j] + bv;
                size_t idx = (size_t)rowg * Nn + colg;
                if (outF) outF[idx] = resid[idx] + g * val;
                else      outBf[idx] = f2bf(val);
            }
        }
    }
}

// ---------------------------------------------------------------------------
// Flash-style masked cross-attention: per (b, h, 128 L rows), loop N in 64-col
// chunks (double-buffered K/V staging). S = Q K^T via WMMA, online softmax,
// O += P V via WMMA.
// ---------------------------------------------------------------------------
__global__ __launch_bounds__(256) void xattn_attention(
    const unsigned short* __restrict__ Qb,
    const unsigned short* __restrict__ Kb,
    const unsigned short* __restrict__ Vb,
    const int* __restrict__ maskp,
    unsigned short* __restrict__ Ob)
{
    __shared__ unsigned short Ks[2][64 * 72];     // [n][64+8] row-major
    __shared__ unsigned short Vt[2][64 * 72];     // [d][64+8] transposed
    __shared__ unsigned short Ps[8 * 16 * 72];    // per-wave P scratch [16][72]

    const int tid  = threadIdx.x;
    const int wid  = tid >> 5, lane = tid & 31;
    const int half = lane >> 4, l16 = lane & 15;
    const int b = blockIdx.z, h = blockIdx.y;
    const int l0 = blockIdx.x * 128 + wid * 16;

    // Q fragments for this wave's 16 rows (reused across all N chunks)
    FragA qf[2];
    {
        const unsigned int* qp = (const unsigned int*)
            (Qb + ((size_t)(b * 2048 + l0 + l16)) * 1024 + h * 64);
        #pragma unroll
        for (int ks = 0; ks < 2; ks++)
            #pragma unroll
            for (int j = 0; j < 8; j++) {
                int pair = ks * 16 + ((j < 4) ? (j + 4 * half)
                                              : (8 + (j - 4) + 4 * half));
                qf[ks].u[j] = qp[pair];
            }
    }

    auto stageKV = [&](int buf, int n0) {
        unsigned long long kbase = (unsigned long long)(uintptr_t)
            (Kb + ((size_t)(b * 1024 + n0)) * 1024 + h * 64);
        #pragma unroll
        for (int t = 0; t < 2; ++t) {
            int q = tid + t * 256;
            int r = q >> 3, seg = q & 7;
            async_b128_s(&Ks[buf][r * 72 + seg * 8], kbase,
                         (unsigned)((r * 1024 + seg * 8) * 2));
        }
        #pragma unroll
        for (int t = 0; t < 2; ++t) {
            int q = tid + t * 256;
            int r = q >> 3, seg = q & 7;
            uint4 d = *(const uint4*)
                (Vb + ((size_t)(b * 1024 + n0 + r)) * 1024 + h * 64 + seg * 8);
            unsigned int w[4] = {d.x, d.y, d.z, d.w};
            #pragma unroll
            for (int e = 0; e < 4; e++) {
                int c = seg * 8 + e * 2;
                Vt[buf][(c + 0) * 72 + r] = (unsigned short)(w[e] & 0xFFFFu);
                Vt[buf][(c + 1) * 72 + r] = (unsigned short)(w[e] >> 16);
            }
        }
    };

    const v8f vzero = {};
    float Mrow[8], Lrow[8];
    v8f oacc[4];
    #pragma unroll
    for (int j = 0; j < 8; j++) { Mrow[j] = -__builtin_inff(); Lrow[j] = 0.f; }
    #pragma unroll
    for (int t = 0; t < 4; t++) oacc[t] = vzero;

    unsigned short* Pw = &Ps[wid * 16 * 72];
    const unsigned int* Pw32 = (const unsigned int*)Pw;

    stageKV(0, 0);

    for (int i = 0; i < 16; ++i) {
        const int p = i & 1;
        const int n0 = i * 64;
        wait_async();
        __syncthreads();                 // buf[p] ready; buf[1-p] free to refill
        if (i + 1 < 16) stageKV(1 - p, (i + 1) * 64);

        const unsigned int* Ks32 = (const unsigned int*)Ks[p];
        const unsigned int* Vt32 = (const unsigned int*)Vt[p];

        // ---- S = Q K^T  (4 n-subtiles, contraction over d = 2 k-steps) ----
        v8f sacc[4];
        #pragma unroll
        for (int t = 0; t < 4; t++) sacc[t] = vzero;
        #pragma unroll
        for (int ks = 0; ks < 2; ks++) {
            #pragma unroll
            for (int nt = 0; nt < 4; nt++) {
                FragA kf;
                #pragma unroll
                for (int j = 0; j < 8; j++)
                    kf.u[j] = Ks32[(nt * 16 + l16) * 36 + ks * 16 + ((j < 4) ? (j + 4 * half) : (4 + j + 4 * half))];
                sacc[nt] = wmma_bf16(qf[ks], kf, sacc[nt]);
            }
        }

        // ---- mask + online softmax ----
        int mk[4];
        #pragma unroll
        for (int nt = 0; nt < 4; nt++)
            mk[nt] = maskp[b * 1024 + n0 + nt * 16 + l16];

        #pragma unroll
        for (int j = 0; j < 8; j++) {
            float m = -__builtin_inff();
            #pragma unroll
            for (int nt = 0; nt < 4; nt++) {
                float s = mk[nt] ? sacc[nt][j] * 0.125f : -__builtin_inff();
                sacc[nt][j] = s;
                m = fmaxf(m, s);
            }
            #pragma unroll
            for (int o = 1; o < 16; o <<= 1) m = fmaxf(m, __shfl_xor(m, o, 32));

            float Mn = fmaxf(Mrow[j], m);
            float alpha = (Mn == -__builtin_inff()) ? 1.f : __expf(Mrow[j] - Mn);
            float lsum = 0.f;
            float pv[4];
            #pragma unroll
            for (int nt = 0; nt < 4; nt++) {
                float s = sacc[nt][j];
                pv[nt] = (s == -__builtin_inff()) ? 0.f : __expf(s - Mn);
                lsum += pv[nt];
            }
            #pragma unroll
            for (int o = 1; o < 16; o <<= 1) lsum += __shfl_xor(lsum, o, 32);
            Lrow[j] = Lrow[j] * alpha + lsum;
            Mrow[j] = Mn;
            #pragma unroll
            for (int t = 0; t < 4; t++) oacc[t][j] *= alpha;
            // P -> per-wave LDS (C layout row j+8*half -> A-frag layout)
            #pragma unroll
            for (int nt = 0; nt < 4; nt++)
                Pw[(j + 8 * half) * 72 + nt * 16 + l16] = f2bf(pv[nt]);
        }
        asm volatile("" ::: "memory");
        wait_ds();                        // per-wave P scratch: RAW through LDS

        // ---- O += P @ V_chunk ----
        #pragma unroll
        for (int ks = 0; ks < 2; ks++) {
            FragA pf;
            #pragma unroll
            for (int j = 0; j < 8; j++) {
                int pair = ks * 16 + ((j < 4) ? (j + 4 * half)
                                              : (8 + (j - 4) + 4 * half));
                pf.u[j] = Pw32[l16 * 36 + pair];
            }
            #pragma unroll
            for (int dt = 0; dt < 4; dt++) {
                FragA vf;
                #pragma unroll
                for (int j = 0; j < 8; j++)
                    vf.u[j] = Vt32[(dt * 16 + l16) * 36 + ks * 16 + ((j < 4) ? (j + 4 * half) : (4 + j + 4 * half))];
                oacc[dt] = wmma_bf16(pf, vf, oacc[dt]);
            }
        }
    }

    // finalize: divide by row sums, store bf16
    #pragma unroll
    for (int j = 0; j < 8; j++) {
        float inv = (Lrow[j] > 0.f) ? 1.f / Lrow[j] : 0.f;
        int l = l0 + j + 8 * half;
        #pragma unroll
        for (int dt = 0; dt < 4; dt++)
            Ob[((size_t)(b * 2048 + l)) * 1024 + h * 64 + dt * 16 + l16] =
                f2bf(oacc[dt][j] * inv);
    }
}

// ---------------------------------------------------------------------------
// Host-side launch
// ---------------------------------------------------------------------------
extern "C" void kernel_launch(void* const* d_in, const int* in_sizes, int n_in,
                              void* d_out, int out_size, void* d_ws, size_t ws_size,
                              hipStream_t stream) {
    (void)in_sizes; (void)n_in; (void)out_size; (void)ws_size;
    constexpr int Bb = 4, Ll = 2048, Nn = 1024, DM = 1024, DN = 512, Hh = 16;

    const float* text_h = (const float*)d_in[0];
    const float* nar_h  = (const float*)d_in[1];
    const int*   nmask  = (const int*)  d_in[2];
    const float* Wq = (const float*)d_in[3];  const float* bq = (const float*)d_in[4];
    const float* Wk = (const float*)d_in[5];  const float* bk = (const float*)d_in[6];
    const float* Wv = (const float*)d_in[7];  const float* bv = (const float*)d_in[8];
    const float* Wo = (const float*)d_in[9];  const float* bo = (const float*)d_in[10];
    const float* gate = (const float*)d_in[11];
    float* out = (float*)d_out;

    const size_t nText = (size_t)Bb * Ll * DM;   // 8M
    const size_t nNar  = (size_t)Bb * Nn * DN;   // 2M
    const size_t nWq = (size_t)DM * DM, nWk = (size_t)DN * DM;
    const size_t nWv = nWk, nWo = nWq;

    char* ws = (char*)d_ws;
    size_t off = 0;
    auto carve = [&](size_t elems) { unsigned short* p = (unsigned short*)(ws + off);
                                     off += elems * 2; return p; };
    unsigned short* text_bf = carve(nText);
    unsigned short* nar_bf  = carve(nNar);
    unsigned short* wq_bf   = carve(nWq);
    unsigned short* wk_bf   = carve(nWk);
    unsigned short* wv_bf   = carve(nWv);
    unsigned short* wo_bf   = carve(nWo);
    unsigned short* q_bf    = carve((size_t)Bb * Ll * DM);
    unsigned short* k_bf    = carve((size_t)Bb * Nn * DM);
    unsigned short* v_bf    = carve((size_t)Bb * Nn * DM);
    unsigned short* o_bf    = carve((size_t)Bb * Ll * DM);

    // 1) fp32 -> bf16 conversions (vectorized x4)
    auto cvt = [&](const float* src, unsigned short* dst, size_t n, int blocks) {
        xattn_cvt_bf16<<<blocks, 256, 0, stream>>>(
            (const float4*)src, (ushort4*)dst, (int)(n >> 2));
    };
    cvt(text_h, text_bf, nText, 2048);
    cvt(nar_h,  nar_bf,  nNar,  1024);
    cvt(Wq, wq_bf, nWq, 512);
    cvt(Wk, wk_bf, nWk, 256);
    cvt(Wv, wv_bf, nWv, 256);
    cvt(Wo, wo_bf, nWo, 512);

    // 2) projections (WMMA GEMMs)
    xattn_gemm_bf16<<<dim3(DM / 128, (Bb * Ll) / 128), 256, 0, stream>>>(
        text_bf, wq_bf, bq, q_bf, nullptr, nullptr, nullptr, Bb * Ll, DM, DM);
    xattn_gemm_bf16<<<dim3(DM / 128, (Bb * Nn) / 128), 256, 0, stream>>>(
        nar_bf, wk_bf, bk, k_bf, nullptr, nullptr, nullptr, Bb * Nn, DN, DM);
    xattn_gemm_bf16<<<dim3(DM / 128, (Bb * Nn) / 128), 256, 0, stream>>>(
        nar_bf, wv_bf, bv, v_bf, nullptr, nullptr, nullptr, Bb * Nn, DN, DM);

    // 3) masked flash attention
    xattn_attention<<<dim3(Ll / 128, Hh, Bb), 256, 0, stream>>>(
        q_bf, k_bf, v_bf, nmask, o_bf);

    // 4) output projection + gated residual (fp32 epilogue)
    xattn_gemm_bf16<<<dim3(DM / 128, (Bb * Ll) / 128), 256, 0, stream>>>(
        o_bf, wo_bf, bo, nullptr, out, text_h, gate, Bb * Ll, DM, DM);
}
